// CausalSelfAttention_39144331936112
// MI455X (gfx1250) — matrix operand order repaired and checked by reading, not compiled, not run on hardware
//
#include <hip/hip_runtime.h>
#include <hip/hip_bf16.h>
#include <math.h>
#include <stdint.h>

typedef __bf16 bf16;
typedef __attribute__((ext_vector_type(16))) __bf16 v16bf;
typedef __attribute__((ext_vector_type(8)))  __bf16 v8bf;
typedef __attribute__((ext_vector_type(4)))  __bf16 v4bf;
typedef __attribute__((ext_vector_type(8)))  float  v8f;

#define B_SZ 4
#define T_SZ 2048
#define C_SZ 1024
#define NH_SZ 16
#define HS_SZ 64
#define M_SZ (B_SZ * T_SZ)   // 8192

#define BK  64               // K-step per stage = 2 WMMA K-steps
#define LST 80               // LDS row stride (bf16 units): 160B rows, 16B aligned

union V16U { v16bf v; v8bf h[2]; };

// --- CDNA5 async global->LDS path (ASYNCcnt), with portable fallback --------
#if defined(__gfx1250__) && __has_builtin(__builtin_amdgcn_global_load_async_to_lds_b128) && __has_builtin(__builtin_amdgcn_s_wait_asynccnt)
#define USE_ASYNC_LDS 1
#else
#define USE_ASYNC_LDS 0
#endif

#if USE_ASYNC_LDS
// Builtin signature (from compiler diagnostic): param0 = int4* in global AS.
typedef int v4i_ __attribute__((vector_size(16)));
typedef __attribute__((address_space(1))) v4i_ g_v4i;
typedef __attribute__((address_space(3))) v4i_ s_v4i;
__device__ __forceinline__ void async_copy_b128(const bf16* g, bf16* l) {
  // Generic LDS pointer low 32 bits == LDS byte address (ISA aperture rule).
  __builtin_amdgcn_global_load_async_to_lds_b128(
      (g_v4i*)(uintptr_t)g, (s_v4i*)(uint32_t)(uintptr_t)l, 0, 0);
}
#endif

__device__ __forceinline__ v8f v8f_zero() {
  v8f z;
#pragma unroll
  for (int i = 0; i < 8; ++i) z[i] = 0.f;
  return z;
}

// ---------------------------------------------------------------------------
// FP32 -> BF16 cast (vectorized 4-wide)
// ---------------------------------------------------------------------------
__global__ void cast_f32_bf16_kernel(const float* __restrict__ in,
                                     bf16* __restrict__ out, int n4) {
  int i = blockIdx.x * blockDim.x + threadIdx.x;
  int stride = gridDim.x * blockDim.x;
  for (; i < n4; i += stride) {
    float4 f = ((const float4*)in)[i];
    v4bf o;
    o[0] = (bf16)f.x; o[1] = (bf16)f.y; o[2] = (bf16)f.z; o[3] = (bf16)f.w;
    ((v4bf*)out)[i] = o;
  }
}

// ---------------------------------------------------------------------------
// Stage one 128x64 bf16 tile of X and of W into LDS (16 x b128 per thread).
// ---------------------------------------------------------------------------
__device__ __forceinline__ void stage_tile(const bf16* __restrict__ X,
                                           const bf16* __restrict__ W,
                                           bf16* As, bf16* Bs,
                                           int mB, int nB, int k0, int tid) {
#if USE_ASYNC_LDS
#pragma unroll
  for (int p = 0; p < 8; ++p) {
    int id  = p * 128 + tid;
    int row = id >> 3;
    int c8  = (id & 7) * 8;
    async_copy_b128(X + (size_t)(mB + row) * C_SZ + k0 + c8, &As[row * LST + c8]);
    async_copy_b128(W + (size_t)(nB + row) * C_SZ + k0 + c8, &Bs[row * LST + c8]);
  }
#else
  uint4 tA[8], tB[8];
#pragma unroll
  for (int p = 0; p < 8; ++p) {
    int id  = p * 128 + tid;
    int row = id >> 3;
    int c8  = (id & 7) * 8;
    tA[p] = *(const uint4*)(X + (size_t)(mB + row) * C_SZ + k0 + c8);
    tB[p] = *(const uint4*)(W + (size_t)(nB + row) * C_SZ + k0 + c8);
  }
#pragma unroll
  for (int p = 0; p < 8; ++p) {
    int id  = p * 128 + tid;
    int row = id >> 3;
    int c8  = (id & 7) * 8;
    *(uint4*)(&As[row * LST + c8]) = tA[p];
    *(uint4*)(&Bs[row * LST + c8]) = tB[p];
  }
#endif
}

// ---------------------------------------------------------------------------
// Double-buffered BF16 GEMM core: acc += X(M x K) * W(N x K)^T block tile.
// Block: 128 threads = 4 waves (2x2 wave grid of 64x64 tiles).
// ---------------------------------------------------------------------------
__device__ __forceinline__ void gemm_core(const bf16* __restrict__ X,
                                          const bf16* __restrict__ W,
                                          bf16* As, bf16* Bs,  // [2][128*LST]
                                          int mB, int nB, v8f acc[4][4]) {
  const int tid  = threadIdx.x;
  const int lane = tid & 31;
  const int wv   = tid >> 5;
  const int wm   = (wv >> 1) * 64;
  const int wn   = (wv & 1) * 64;
  const int hf   = lane >> 4;
  const int r    = lane & 15;

  stage_tile(X, W, As, Bs, mB, nB, 0, tid);

  for (int k0 = 0; k0 < C_SZ; k0 += BK) {
    const int cur = (k0 >> 6) & 1;
    bf16* curA = As + cur * (128 * LST);
    bf16* curB = Bs + cur * (128 * LST);

    if (k0 + BK < C_SZ) {
      // prefetch next K-slab into the other buffer while we compute
      stage_tile(X, W, As + (cur ^ 1) * (128 * LST), Bs + (cur ^ 1) * (128 * LST),
                 mB, nB, k0 + BK, tid);
#if USE_ASYNC_LDS
      __builtin_amdgcn_s_wait_asynccnt(16);  // drain previous stage (in-order)
#endif
    } else {
#if USE_ASYNC_LDS
      __builtin_amdgcn_s_wait_asynccnt(0);
#endif
    }
    if (k0 + 2 * BK < C_SZ)
      __builtin_prefetch(X + (size_t)(mB + tid) * C_SZ + k0 + 2 * BK, 0, 0);
    __syncthreads();

#pragma unroll
    for (int ks = 0; ks < 2; ++ks) {
      v16bf af[4], bfr[4];
#pragma unroll
      for (int mi = 0; mi < 4; ++mi) {
        int row = wm + mi * 16 + r;
        V16U u;
        u.h[0] = *(const v8bf*)(&curA[row * LST + ks * 32 + hf * 8]);
        u.h[1] = *(const v8bf*)(&curA[row * LST + ks * 32 + 16 + hf * 8]);
        af[mi] = u.v;
      }
#pragma unroll
      for (int ni = 0; ni < 4; ++ni) {
        int row = wn + ni * 16 + r;
        V16U u;
        u.h[0] = *(const v8bf*)(&curB[row * LST + ks * 32 + hf * 8]);
        u.h[1] = *(const v8bf*)(&curB[row * LST + ks * 32 + 16 + hf * 8]);
        bfr[ni] = u.v;
      }
#pragma unroll
      for (int mi = 0; mi < 4; ++mi)
#pragma unroll
        for (int ni = 0; ni < 4; ++ni)
          acc[mi][ni] = __builtin_amdgcn_wmma_f32_16x16x32_bf16(
              false, af[mi], false, bfr[ni], (short)0, acc[mi][ni], false, false);
    }
    __syncthreads();  // protect cur buffer before it is restaged
  }
}

// ---------------------------------------------------------------------------
// QKV projection GEMM. mode 0: out[b,h,t,d] (Q,K). mode 1: out[b,h,d,t] (V^T)
// ---------------------------------------------------------------------------
__global__ __launch_bounds__(128) void gemm_qkv_kernel(const bf16* __restrict__ X,
                                                       const bf16* __restrict__ W,
                                                       bf16* __restrict__ out,
                                                       int mode) {
  __shared__ bf16 As[2 * 128 * LST];
  __shared__ bf16 Bs[2 * 128 * LST];
  const int lane = threadIdx.x & 31;
  const int wv   = threadIdx.x >> 5;
  const int wm   = (wv >> 1) * 64;
  const int wn   = (wv & 1) * 64;
  const int hf   = lane >> 4;
  const int r    = lane & 15;
  const int mB   = blockIdx.y * 128;
  const int nB   = blockIdx.x * 128;

  v8f acc[4][4];
#pragma unroll
  for (int i = 0; i < 4; ++i)
#pragma unroll
    for (int j = 0; j < 4; ++j) acc[i][j] = v8f_zero();

  gemm_core(X, W, As, Bs, mB, nB, acc);

#pragma unroll
  for (int mi = 0; mi < 4; ++mi)
#pragma unroll
    for (int ni = 0; ni < 4; ++ni)
#pragma unroll
      for (int e = 0; e < 8; ++e) {
        int m = mB + wm + mi * 16 + hf * 8 + e;
        int n = nB + wn + ni * 16 + r;
        int b = m >> 11, t = m & (T_SZ - 1);
        int h = n >> 6,  d = n & (HS_SZ - 1);
        bf16 val = (bf16)acc[mi][ni][e];
        if (mode == 0)
          out[((size_t)(b * NH_SZ + h) * T_SZ + t) * HS_SZ + d] = val;
        else
          out[((size_t)(b * NH_SZ + h) * HS_SZ + d) * T_SZ + t] = val;
      }
}

// ---------------------------------------------------------------------------
// Final projection: out(M x N, fp32) = Y(M x K) * Wo(N x K)^T + bias
// ---------------------------------------------------------------------------
__global__ __launch_bounds__(128) void gemm_out_kernel(const bf16* __restrict__ Y,
                                                       const bf16* __restrict__ W,
                                                       const float* __restrict__ bias,
                                                       float* __restrict__ out) {
  __shared__ bf16 As[2 * 128 * LST];
  __shared__ bf16 Bs[2 * 128 * LST];
  const int lane = threadIdx.x & 31;
  const int wv   = threadIdx.x >> 5;
  const int wm   = (wv >> 1) * 64;
  const int wn   = (wv & 1) * 64;
  const int hf   = lane >> 4;
  const int r    = lane & 15;
  const int mB   = blockIdx.y * 128;
  const int nB   = blockIdx.x * 128;

  v8f acc[4][4];
#pragma unroll
  for (int i = 0; i < 4; ++i)
#pragma unroll
    for (int j = 0; j < 4; ++j) acc[i][j] = v8f_zero();

  gemm_core(Y, W, As, Bs, mB, nB, acc);

#pragma unroll
  for (int mi = 0; mi < 4; ++mi)
#pragma unroll
    for (int ni = 0; ni < 4; ++ni)
#pragma unroll
      for (int e = 0; e < 8; ++e) {
        int m = mB + wm + mi * 16 + hf * 8 + e;
        int n = nB + wn + ni * 16 + r;
        out[(size_t)m * C_SZ + n] = acc[mi][ni][e] + bias[n];
      }
}

// ---------------------------------------------------------------------------
// Flash attention, one wave per (b*h, 64-row Q tile).
// Q,K: (B*NH, T, HS) bf16; Vt: (B*NH, HS, T) bf16; Y: (B, T, C) bf16
// ---------------------------------------------------------------------------
__global__ __launch_bounds__(32) void attn_kernel(const bf16* __restrict__ Q,
                                                  const bf16* __restrict__ K,
                                                  const bf16* __restrict__ Vt,
                                                  bf16* __restrict__ Y) {
  __shared__ bf16 Ps[64 * 72];   // P tile staging (C-layout -> A-layout)
  const int lane = threadIdx.x;
  const int hf = lane >> 4, r = lane & 15;
  const int qt = blockIdx.x;
  const int bh = blockIdx.y;
  const bf16* q  = Q  + (size_t)bh * T_SZ * HS_SZ;
  const bf16* k  = K  + (size_t)bh * T_SZ * HS_SZ;
  const bf16* vt = Vt + (size_t)bh * HS_SZ * T_SZ;

  v16bf qf[4][2];
#pragma unroll
  for (int mi = 0; mi < 4; ++mi)
#pragma unroll
    for (int ks = 0; ks < 2; ++ks) {
      const bf16* base = q + (size_t)(qt * 64 + mi * 16 + r) * HS_SZ + ks * 32;
      V16U u;
      u.h[0] = *(const v8bf*)(base + hf * 8);
      u.h[1] = *(const v8bf*)(base + 16 + hf * 8);
      qf[mi][ks] = u.v;
    }

  float mst[4][8], lst[4][8];
  v8f o[4][4];
#pragma unroll
  for (int mi = 0; mi < 4; ++mi) {
#pragma unroll
    for (int e = 0; e < 8; ++e) { mst[mi][e] = -3.0e38f; lst[mi][e] = 0.f; }
#pragma unroll
    for (int ni = 0; ni < 4; ++ni) o[mi][ni] = v8f_zero();
  }
  const float scale = 0.125f;  // 1/sqrt(HS)

  for (int j = 0; j <= qt; ++j) {
    v8f s[4][4];
#pragma unroll
    for (int mi = 0; mi < 4; ++mi)
#pragma unroll
      for (int ni = 0; ni < 4; ++ni) s[mi][ni] = v8f_zero();
#pragma unroll
    for (int ks = 0; ks < 2; ++ks) {
      v16bf kf[4];
#pragma unroll
      for (int ni = 0; ni < 4; ++ni) {
        const bf16* base = k + (size_t)(j * 64 + ni * 16 + r) * HS_SZ + ks * 32;
        V16U u;
        u.h[0] = *(const v8bf*)(base + hf * 8);
        u.h[1] = *(const v8bf*)(base + 16 + hf * 8);
        kf[ni] = u.v;
      }
#pragma unroll
      for (int mi = 0; mi < 4; ++mi)
#pragma unroll
        for (int ni = 0; ni < 4; ++ni)
          s[mi][ni] = __builtin_amdgcn_wmma_f32_16x16x32_bf16(
              false, qf[mi][ks], false, kf[ni], (short)0, s[mi][ni], false, false);
    }

#pragma unroll
    for (int mi = 0; mi < 4; ++mi) {
      float rmax[8];
#pragma unroll
      for (int e = 0; e < 8; ++e) rmax[e] = -3.0e38f;
#pragma unroll
      for (int ni = 0; ni < 4; ++ni)
#pragma unroll
        for (int e = 0; e < 8; ++e) {
          float val = s[mi][ni][e] * scale;
          int kv = j * 64 + ni * 16 + r;
          int qp = qt * 64 + mi * 16 + hf * 8 + e;
          val = (kv <= qp) ? val : -3.0e38f;
          s[mi][ni][e] = val;
          rmax[e] = fmaxf(rmax[e], val);
        }
#pragma unroll
      for (int e = 0; e < 8; ++e)
#pragma unroll
        for (int off = 1; off < 16; off <<= 1)
          rmax[e] = fmaxf(rmax[e], __shfl_xor(rmax[e], off, 32));

      float rsum[8];
#pragma unroll
      for (int e = 0; e < 8; ++e) {
        float mnew = fmaxf(mst[mi][e], rmax[e]);
        float alpha = __expf(mst[mi][e] - mnew);
        mst[mi][e] = mnew;
        lst[mi][e] *= alpha;
        rsum[e] = 0.f;
#pragma unroll
        for (int ni = 0; ni < 4; ++ni) o[mi][ni][e] *= alpha;
      }
#pragma unroll
      for (int ni = 0; ni < 4; ++ni)
#pragma unroll
        for (int e = 0; e < 8; ++e) {
          float p = __expf(s[mi][ni][e] - mst[mi][e]);
          rsum[e] += p;
          Ps[(mi * 16 + hf * 8 + e) * 72 + ni * 16 + r] = (bf16)p;
        }
#pragma unroll
      for (int e = 0; e < 8; ++e) {
#pragma unroll
        for (int off = 1; off < 16; off <<= 1)
          rsum[e] += __shfl_xor(rsum[e], off, 32);
        lst[mi][e] += rsum[e];
      }
    }
    __syncthreads();

#pragma unroll
    for (int ks = 0; ks < 2; ++ks) {
      v16bf pf[4], vf[4];
#pragma unroll
      for (int mi = 0; mi < 4; ++mi) {
        const bf16* base = Ps + (mi * 16 + r) * 72 + ks * 32;
        V16U u;
        u.h[0] = *(const v8bf*)(base + hf * 8);
        u.h[1] = *(const v8bf*)(base + 16 + hf * 8);
        pf[mi] = u.v;
      }
#pragma unroll
      for (int ni = 0; ni < 4; ++ni) {
        const bf16* base = vt + (size_t)(ni * 16 + r) * T_SZ + j * 64 + ks * 32;
        V16U u;
        u.h[0] = *(const v8bf*)(base + hf * 8);
        u.h[1] = *(const v8bf*)(base + 16 + hf * 8);
        vf[ni] = u.v;
      }
#pragma unroll
      for (int mi = 0; mi < 4; ++mi)
#pragma unroll
        for (int ni = 0; ni < 4; ++ni)
          o[mi][ni] = __builtin_amdgcn_wmma_f32_16x16x32_bf16(
              false, pf[mi], false, vf[ni], (short)0, o[mi][ni], false, false);
    }
    __syncthreads();
  }

  int b = bh >> 4, h = bh & 15;
#pragma unroll
  for (int mi = 0; mi < 4; ++mi)
#pragma unroll
    for (int ni = 0; ni < 4; ++ni)
#pragma unroll
      for (int e = 0; e < 8; ++e) {
        int t = qt * 64 + mi * 16 + hf * 8 + e;
        int d = ni * 16 + r;
        float val = o[mi][ni][e] / lst[mi][e];
        Y[((size_t)(b * T_SZ + t)) * C_SZ + h * HS_SZ + d] = (bf16)val;
      }
}

// ---------------------------------------------------------------------------
extern "C" void kernel_launch(void* const* d_in, const int* in_sizes, int n_in,
                              void* d_out, int out_size, void* d_ws, size_t ws_size,
                              hipStream_t stream) {
  const float* x  = (const float*)d_in[0];
  const float* Wk = (const float*)d_in[1];
  const float* Wq = (const float*)d_in[2];
  const float* Wv = (const float*)d_in[3];
  const float* Wo = (const float*)d_in[4];
  const float* bo = (const float*)d_in[5];

  char* ws = (char*)d_ws;
  size_t off = 0;
  auto carve = [&](size_t bytes) {
    char* p = ws + off;
    off += (bytes + 255) & ~(size_t)255;
    return p;
  };
  bf16* xb  = (bf16*)carve((size_t)M_SZ * C_SZ * 2);
  bf16* wqb = (bf16*)carve((size_t)C_SZ * C_SZ * 2);
  bf16* wkb = (bf16*)carve((size_t)C_SZ * C_SZ * 2);
  bf16* wvb = (bf16*)carve((size_t)C_SZ * C_SZ * 2);
  bf16* wob = (bf16*)carve((size_t)C_SZ * C_SZ * 2);
  bf16* qs  = (bf16*)carve((size_t)M_SZ * C_SZ * 2);   // (B,NH,T,HS)
  bf16* ks_ = (bf16*)carve((size_t)M_SZ * C_SZ * 2);   // (B,NH,T,HS)
  bf16* vts = (bf16*)carve((size_t)M_SZ * C_SZ * 2);   // (B,NH,HS,T)
  bf16* ys  = (bf16*)carve((size_t)M_SZ * C_SZ * 2);   // (B,T,C)

  cast_f32_bf16_kernel<<<512, 256, 0, stream>>>(x,  xb,  M_SZ * C_SZ / 4);
  cast_f32_bf16_kernel<<<128, 256, 0, stream>>>(Wq, wqb, C_SZ * C_SZ / 4);
  cast_f32_bf16_kernel<<<128, 256, 0, stream>>>(Wk, wkb, C_SZ * C_SZ / 4);
  cast_f32_bf16_kernel<<<128, 256, 0, stream>>>(Wv, wvb, C_SZ * C_SZ / 4);
  cast_f32_bf16_kernel<<<128, 256, 0, stream>>>(Wo, wob, C_SZ * C_SZ / 4);

  dim3 gg(C_SZ / 128, M_SZ / 128);   // (8, 64)
  gemm_qkv_kernel<<<gg, 128, 0, stream>>>(xb, wqb, qs, 0);
  gemm_qkv_kernel<<<gg, 128, 0, stream>>>(xb, wkb, ks_, 0);
  gemm_qkv_kernel<<<gg, 128, 0, stream>>>(xb, wvb, vts, 1);

  attn_kernel<<<dim3(T_SZ / 64, B_SZ * NH_SZ), 32, 0, stream>>>(qs, ks_, vts, ys);

  gemm_out_kernel<<<gg, 128, 0, stream>>>(ys, wob, bo, (float*)d_out);
}
